// Multi_Head_Attention_71536975282553
// MI455X (gfx1250) — matrix operand order repaired and checked by reading, not compiled, not run on hardware
//
#include <hip/hip_runtime.h>

// ---------------------------------------------------------------------------
// MHA forward for MI455X (gfx1250, wave32, WMMA 16x16x32 bf16 / f32-acc)
// Pipeline: [Q,K,V proj GEMM] -> [flash attention w/ TDM double-buffering]
//           -> [output proj GEMM]
// Matrix math via v_wmma_f32_16x16x32_bf16; K/V tiles staged to LDS by the
// Tensor Data Mover (tensor_load_to_lds) and synchronized with s_wait_tensorcnt.
// ---------------------------------------------------------------------------

#define DMODEL 1024
#define NHEADS 16
#define DK     64
#define BATCH  4
#define SEQ    2048
#define TOKENS (BATCH * SEQ)   // 8192

typedef __attribute__((ext_vector_type(16))) __bf16 v16bf;
typedef __attribute__((ext_vector_type(8)))  __bf16 v8bf;
typedef __attribute__((ext_vector_type(8)))  float  v8f;
typedef __attribute__((ext_vector_type(4)))  unsigned int v4u;
typedef __attribute__((ext_vector_type(8)))  int v8i;
typedef __attribute__((ext_vector_type(4)))  int v4i;

union ABfrag { struct { v8bf lo, hi; } p; v16bf v; };

__device__ __forceinline__ v8f wmma_bf16(v16bf a, v16bf b, v8f c) {
    return __builtin_amdgcn_wmma_f32_16x16x32_bf16(
        /*neg_a=*/false, a, /*neg_b=*/false, b,
        /*c_mod=*/(short)0, c, /*reuse_a=*/false, /*reuse_b=*/false);
}

// ---------------------------------------------------------------------------
// Tensor Data Mover: 2D tile (bf16 elements) global -> LDS.
// D# per CDNA5 ISA: group0 = {flags, lds_addr,
// global_addr[56:0], type=2}; group1 = {data_size=1 (2B), tensor dims,
// tile dims, dim0 stride}; groups 2/3 zero (2D tensor). Tracked by TENSORcnt.
// This toolchain exposes the 6-arg builtin form (extra int32x8 group + cpol).
// ---------------------------------------------------------------------------
__device__ __forceinline__ void tdm_load_2d_bf16(
    unsigned int lds_off, const void* gptr,
    unsigned int tdim0, unsigned int tdim1,
    unsigned int tile0, unsigned int tile1,
    unsigned int stride0 /* elements */)
{
    const unsigned long long ga = (unsigned long long)(size_t)gptr;
    v4u g0;
    g0[0] = 1u;                                              // count=1, no gather
    g0[1] = lds_off;                                         // LDS byte address
    g0[2] = (unsigned int)ga;                                // global_addr[31:0]
    g0[3] = (unsigned int)((ga >> 32) & 0x01FFFFFFu)         // global_addr[56:32]
          | (2u << 30);                                      // type = 2 (image)
    v8i g1;
    g1[0] = (int)(1u << 16);                                 // data_size=1 -> 2B
    g1[1] = (int)((tdim0 & 0xFFFFu) << 16);                  // tensor_dim0[15:0]
    g1[2] = (int)(((tdim0 >> 16) & 0xFFFFu) |
                  ((tdim1 & 0xFFFFu) << 16));                // td0 hi | td1 lo
    g1[3] = (int)(((tdim1 >> 16) & 0xFFFFu) |
                  ((tile0 & 0xFFFFu) << 16));                // td1 hi | tile_dim0
    g1[4] = (int)(tile1 & 0xFFFFu);                          // tile_dim1, tile_dim2=0
    g1[5] = (int)stride0;                                    // tensor_dim0_stride lo
    g1[6] = 0;                                               // stride0 hi | stride1 lo
    g1[7] = 0;
    const v4i z4 = {0, 0, 0, 0};
    const v8i z8 = {0, 0, 0, 0, 0, 0, 0, 0};
    __builtin_amdgcn_tensor_load_to_lds(g0, g1, z4, z4, z8, 0);
}

enum { MODE_BF16 = 0, MODE_VT = 1, MODE_F32 = 2 };

// ---------------------------------------------------------------------------
// Y[M,N] = X[M,K] @ W[N,K]^T + bias    (reference does x @ W.T + b)
// Block tile 64x128, BK=32; 8 waves, each wave computes a 32x32 tile (2x2 WMMA).
// ---------------------------------------------------------------------------
template <typename InT, int MODE>
__global__ __launch_bounds__(256) void gemm_wmma_kernel(
    const InT*  __restrict__ X,
    const float* __restrict__ W,
    const float* __restrict__ bias,
    void* __restrict__ Yv,
    int M, int N, int K)
{
    __shared__ __bf16 As[64][32];    // 4 KB
    __shared__ __bf16 Bs[128][32];   // 8 KB

    const int tid   = threadIdx.x;
    const int wid   = tid >> 5;
    const int lane  = tid & 31;
    const int lk    = lane & 15;
    const int bk0   = (lane < 16) ? 0 : 16;   // B-frag k offset (16 contiguous k)
    const int ak0   = (lane < 16) ? 0 : 16;   // A-frag chunks: k = ak0..+7, ak0+8..+15
    const int half8 = (lane < 16) ? 0 : 8;    // C-layout row offset
    const int wm    = wid >> 2;               // 0..1
    const int wn    = wid & 3;                // 0..3

    const int m0 = blockIdx.y * 64;
    const int n0 = blockIdx.x * 128;

    v8f acc[2][2] = {};

    const int ar = tid >> 2, ac = (tid & 3) * 8;   // A: 64 rows x 8 cols each
    const int br = tid >> 1, bc = (tid & 1) * 16;  // B: 128 rows x 16 cols each

    for (int kk = 0; kk < K; kk += 32) {
        {
            const InT* src = X + (size_t)(m0 + ar) * K + kk + ac;
#pragma unroll
            for (int e = 0; e < 8; e++) As[ar][ac + e] = (__bf16)(float)src[e];
        }
        {
            const float* src = W + (size_t)(n0 + br) * K + kk + bc;
#pragma unroll
            for (int e = 0; e < 16; e++) Bs[br][bc + e] = (__bf16)src[e];
        }
        if (kk + 32 < K) {   // warm L2/WGP$ one k-step ahead (global_prefetch_b8)
            __builtin_prefetch(X + (size_t)(m0 + ar) * K + kk + 32 + ac, 0, 1);
            __builtin_prefetch(W + (size_t)(n0 + br) * K + kk + 32 + bc, 0, 1);
        }
        __syncthreads();

        ABfrag afr[2];
        v16bf bfr[2];
#pragma unroll
        for (int i = 0; i < 2; i++) {
            const __bf16* p = &As[wm * 32 + i * 16 + lk][ak0];
            afr[i].p.lo = *(const v8bf*)(p);
            afr[i].p.hi = *(const v8bf*)(p + 8);
        }
#pragma unroll
        for (int j = 0; j < 2; j++)
            bfr[j] = *(const v16bf*)&Bs[wn * 32 + j * 16 + lk][bk0];

#pragma unroll
        for (int i = 0; i < 2; i++)
#pragma unroll
            for (int j = 0; j < 2; j++)
                acc[i][j] = wmma_bf16(afr[i].v, bfr[j], acc[i][j]);
        __syncthreads();
    }

    // epilogue: C layout  row = r + half8, col = lane&15
#pragma unroll
    for (int i = 0; i < 2; i++) {
#pragma unroll
        for (int j = 0; j < 2; j++) {
#pragma unroll
            for (int r = 0; r < 8; r++) {
                const int m = m0 + wm * 32 + i * 16 + r + half8;
                const int n = n0 + wn * 32 + j * 16 + lk;
                const float val = acc[i][j][r] + bias[n];
                if (MODE == MODE_BF16) {
                    ((__bf16*)Yv)[(size_t)m * N + n] = (__bf16)val;
                } else if (MODE == MODE_F32) {
                    ((float*)Yv)[(size_t)m * N + n] = val;
                } else { // MODE_VT: store V transposed per head: [b, h, d, s]
                    const int b = m / SEQ, s = m % SEQ;
                    const int h = n >> 6, d = n & 63;
                    ((__bf16*)Yv)[((size_t)(b * NHEADS + h) * DK + d) * SEQ + s] = (__bf16)val;
                }
            }
        }
    }
}

// ---------------------------------------------------------------------------
// Flash attention. Grid: (SEQ/128, BATCH*NHEADS); 8 waves/block, 16 q-rows/wave.
// Qh, Kh: bf16 [b, s, h, d]; Vt: bf16 [b, h, d, s]; ctx: bf16 [token, DMODEL].
// K/V tiles for the whole workgroup are staged into a double-buffered LDS
// pipeline by the TDM (wave 0 issues, all 8 waves consume) -> 8x less HBM
// traffic than per-wave global loads, DMA overlapped with WMMA.
// ---------------------------------------------------------------------------
__global__ __launch_bounds__(256) void attn_kernel(
    const __bf16* __restrict__ Qh,
    const __bf16* __restrict__ Kh,
    const __bf16* __restrict__ Vt,
    __bf16* __restrict__ ctx)
{
    __shared__ __bf16 Kbuf[2][32][64];   // 8 KB: 32 keys x 64 d per buffer
    __shared__ __bf16 Vbuf[2][64][32];   // 8 KB: 64 d x 32 keys per buffer
    __shared__ __bf16 Pbuf[8][16][32];   // 8 KB: per-wave P staging (C->A layout)

    const int wid   = threadIdx.x >> 5;
    const int lane  = threadIdx.x & 31;
    const int lk    = lane & 15;
    const int bk0   = (lane < 16) ? 0 : 16;
    const int ak0   = (lane < 16) ? 0 : 16;
    const int half8 = (lane < 16) ? 0 : 8;

    const int bh = blockIdx.y;
    const int b  = bh / NHEADS, h = bh % NHEADS;
    const int q0 = blockIdx.x * 128 + wid * 16;

    const float scale = 0.125f;  // 1/sqrt(64)

    // Q A-fragments (lane row = lk), d split 0..31 / 32..63
    const __bf16* qrow = Qh + (size_t)(b * SEQ + q0 + lk) * DMODEL + h * DK;
    ABfrag qa0, qa1;
    qa0.p.lo = *(const v8bf*)(qrow + ak0);
    qa0.p.hi = *(const v8bf*)(qrow + ak0 + 8);
    qa1.p.lo = *(const v8bf*)(qrow + 32 + ak0);
    qa1.p.hi = *(const v8bf*)(qrow + 32 + ak0 + 8);

    v8f acc[4] = {};                 // out tiles: d = nt*16 + lk
    float mrow[8], lrow[8];
#pragma unroll
    for (int r = 0; r < 8; r++) { mrow[r] = -3.0e38f; lrow[r] = 0.0f; }

    const __bf16* Kgbl = Kh + (size_t)(b * SEQ) * DMODEL + h * DK;
    const __bf16* Vgbl = Vt + (size_t)bh * DK * SEQ;

    // prologue: wave 0 kicks off the first K/V tile pair via TDM
    if (wid == 0) {
        tdm_load_2d_bf16((unsigned int)(size_t)&Kbuf[0][0][0], Kgbl,
                         DK, 32, DK, 32, DMODEL);
        tdm_load_2d_bf16((unsigned int)(size_t)&Vbuf[0][0][0], Vgbl,
                         32, DK, 32, DK, SEQ);
    }

    for (int kb = 0; kb < SEQ; kb += 32) {
        const int cur = (kb >> 5) & 1;

        if (wid == 0) __builtin_amdgcn_s_wait_tensorcnt(0);  // buf[cur] landed
        __syncthreads();                                     // visible to all waves
        if (wid == 0 && kb + 32 < SEQ) {                     // prefetch next tiles
            tdm_load_2d_bf16((unsigned int)(size_t)&Kbuf[cur ^ 1][0][0],
                             Kgbl + (size_t)(kb + 32) * DMODEL,
                             DK, 32, DK, 32, DMODEL);
            tdm_load_2d_bf16((unsigned int)(size_t)&Vbuf[cur ^ 1][0][0],
                             Vgbl + (kb + 32),
                             32, DK, 32, DK, SEQ);
        }

        // ---- scores: two 16x16 tiles, each K-dim = d_k = 64 (2 WMMAs) ----
        v8f s0 = {}, s1 = {};
        {
            v16bf kb0 = *(const v16bf*)&Kbuf[cur][lk][bk0];
            v16bf kb1 = *(const v16bf*)&Kbuf[cur][lk][32 + bk0];
            s0 = wmma_bf16(qa0.v, kb0, s0);
            s0 = wmma_bf16(qa1.v, kb1, s0);
        }
        {
            v16bf kb0 = *(const v16bf*)&Kbuf[cur][16 + lk][bk0];
            v16bf kb1 = *(const v16bf*)&Kbuf[cur][16 + lk][32 + bk0];
            s1 = wmma_bf16(qa0.v, kb0, s1);
            s1 = wmma_bf16(qa1.v, kb1, s1);
        }

        // ---- online softmax over this 32-key block ----
#pragma unroll
        for (int r = 0; r < 8; r++) {
            float v0 = s0[r] * scale;
            float v1 = s1[r] * scale;
            float tm = fmaxf(v0, v1);
            // reduce over 16 keys (lanes within half-wave; xor<16 stays in half)
            tm = fmaxf(tm, __shfl_xor(tm, 1, 32));
            tm = fmaxf(tm, __shfl_xor(tm, 2, 32));
            tm = fmaxf(tm, __shfl_xor(tm, 4, 32));
            tm = fmaxf(tm, __shfl_xor(tm, 8, 32));
            const float nm = fmaxf(mrow[r], tm);
            const float p0 = __expf(v0 - nm);
            const float p1 = __expf(v1 - nm);
            float ts = p0 + p1;
            ts += __shfl_xor(ts, 1, 32);
            ts += __shfl_xor(ts, 2, 32);
            ts += __shfl_xor(ts, 4, 32);
            ts += __shfl_xor(ts, 8, 32);
            const float alpha = __expf(mrow[r] - nm);
            lrow[r] = lrow[r] * alpha + ts;
            mrow[r] = nm;
#pragma unroll
            for (int nt = 0; nt < 4; nt++) acc[nt][r] *= alpha;
            // stage P (C-layout -> LDS row-major 16x32)
            Pbuf[wid][r + half8][lk]      = (__bf16)p0;
            Pbuf[wid][r + half8][16 + lk] = (__bf16)p1;
        }
        __syncthreads();   // Pbuf ready (and keeps waves in lockstep on Kbuf/Vbuf)

        // ---- P as A-fragment (16x32 over keys) ----
        ABfrag pa;
        {
            const __bf16* prow = &Pbuf[wid][lk][0];
            pa.p.lo = *(const v8bf*)(prow + ak0);
            pa.p.hi = *(const v8bf*)(prow + ak0 + 8);
        }

        // ---- out += P @ V_block : 4 d-tiles, B-frags contiguous in LDS ----
#pragma unroll
        for (int nt = 0; nt < 4; nt++) {
            v16bf vb = *(const v16bf*)&Vbuf[cur][nt * 16 + lk][bk0];
            acc[nt] = wmma_bf16(pa.v, vb, acc[nt]);
        }
        __syncthreads();   // all waves done with buf[cur] before TDM refills it
    }

    // ---- finalize + write ctx [token, DMODEL] ----
#pragma unroll
    for (int r = 0; r < 8; r++) {
        const float invl = 1.0f / lrow[r];
        const int m = q0 + r + half8;
#pragma unroll
        for (int nt = 0; nt < 4; nt++) {
            const int chan = h * DK + nt * 16 + lk;
            ctx[(size_t)(b * SEQ + m) * DMODEL + chan] = (__bf16)(acc[nt][r] * invl);
        }
    }
}

// ---------------------------------------------------------------------------
extern "C" void kernel_launch(void* const* d_in, const int* in_sizes, int n_in,
                              void* d_out, int out_size, void* d_ws, size_t ws_size,
                              hipStream_t stream)
{
    (void)in_sizes; (void)n_in; (void)out_size; (void)ws_size;

    const float* q  = (const float*)d_in[0];
    const float* k  = (const float*)d_in[1];
    const float* v  = (const float*)d_in[2];
    const float* Wq = (const float*)d_in[3];
    const float* bq = (const float*)d_in[4];
    const float* Wk = (const float*)d_in[5];
    const float* bk = (const float*)d_in[6];
    const float* Wv = (const float*)d_in[7];
    const float* bv = (const float*)d_in[8];
    const float* Wo = (const float*)d_in[9];
    const float* bo = (const float*)d_in[10];
    float* out = (float*)d_out;

    // workspace: 4 x 16 MB bf16 buffers
    __bf16* Qh  = (__bf16*)d_ws;
    __bf16* Khd = Qh  + (size_t)TOKENS * DMODEL;
    __bf16* Vt  = Khd + (size_t)TOKENS * DMODEL;
    __bf16* ctx = Vt  + (size_t)TOKENS * DMODEL;

    const dim3 gblock(256);
    const dim3 ggrid(DMODEL / 128, TOKENS / 64);   // (8, 128)

    gemm_wmma_kernel<float, MODE_BF16><<<ggrid, gblock, 0, stream>>>(
        q, Wq, bq, Qh, TOKENS, DMODEL, DMODEL);
    gemm_wmma_kernel<float, MODE_BF16><<<ggrid, gblock, 0, stream>>>(
        k, Wk, bk, Khd, TOKENS, DMODEL, DMODEL);
    gemm_wmma_kernel<float, MODE_VT><<<ggrid, gblock, 0, stream>>>(
        v, Wv, bv, Vt, TOKENS, DMODEL, DMODEL);

    attn_kernel<<<dim3(SEQ / 128, BATCH * NHEADS), gblock, 0, stream>>>(Qh, Khd, Vt, ctx);

    gemm_wmma_kernel<__bf16, MODE_F32><<<ggrid, gblock, 0, stream>>>(
        ctx, Wo, bo, out, TOKENS, DMODEL, DMODEL);
}
